// MPNNModel_55207509623274
// MI455X (gfx1250) — compile-verified, matrix operand order checked by reading, not run on hardware
//
#include <hip/hip_runtime.h>
#include <hip/hip_bf16.h>

typedef _Float16 v16h __attribute__((ext_vector_type(16)));
typedef _Float16 v8h  __attribute__((ext_vector_type(8)));
typedef _Float16 v4h  __attribute__((ext_vector_type(4)));
typedef float    v8f  __attribute__((ext_vector_type(8)));
typedef int      v4i  __attribute__((ext_vector_type(4)));

#define DEVINL __device__ __forceinline__

// ---- problem sizes ----
static constexpr int NN = 50000;     // nodes
static constexpr int NE = 600000;    // edges
static constexpr int ND = 64;        // node feature dim
static constexpr int ED = 16;        // edge feature dim
static constexpr int H  = 128;       // hidden
static constexpr int NL = 3;         // layers
static constexpr int NG = 256;       // graphs
static constexpr int NC = 2;         // classes

static constexpr int E_TILES = (NE + 15) / 16;   // 37500 (exact)
static constexpr int N_TILES = (NN + 15) / 16;   // 3125  (exact)
static constexpr unsigned MAXENC_INIT = 0x007FFFFFu;   // enc(-inf)

// ---------------- gfx1250 async global->LDS copy (ASYNCcnt-tracked) ----------------

#if defined(__has_builtin)
#if __has_builtin(__builtin_amdgcn_global_load_async_to_lds_b128) && \
    __has_builtin(__builtin_amdgcn_s_wait_asynccnt)
#define HAVE_ASYNC_LDS 1
#else
#define HAVE_ASYNC_LDS 0
#endif
#else
#define HAVE_ASYNC_LDS 0
#endif

#if HAVE_ASYNC_LDS
typedef __attribute__((address_space(1))) v4i gvec_t;   // global int4
typedef __attribute__((address_space(3))) v4i lvec_t;   // LDS int4
#endif

// copy 16 bytes global -> LDS (per-lane); completion via ASYNCcnt
DEVINL void async_cp16(const void* g, void* l) {
#if HAVE_ASYNC_LDS
  gvec_t* gp = (gvec_t*)(unsigned long long)(uintptr_t)g;
  lvec_t* lp = (lvec_t*)(unsigned int)(uintptr_t)l;
  __builtin_amdgcn_global_load_async_to_lds_b128(gp, lp, 0, 0);
#else
  *(v8h*)l = *(const v8h*)g;
#endif
}

DEVINL void async_wait0() {
#if HAVE_ASYNC_LDS
  __builtin_amdgcn_s_wait_asynccnt(0);
#endif
}

// ---------------- WMMA helpers ----------------

DEVINL v8f wmma_f16(v16h a, v16h b, v8f c) {
  // D = A(16x32 f16) * B(32x16 f16) + C(16x16 f32)
  return __builtin_amdgcn_wmma_f32_16x16x32_f16(false, a, false, b, (short)0, c,
                                                false, false);
}

DEVINL v8f zero8f() {
  v8f z;
#pragma unroll
  for (int i = 0; i < 8; ++i) z[i] = 0.f;
  return z;
}

DEVINL v8h zero8h() {
  v8h z;
#pragma unroll
  for (int i = 0; i < 8; ++i) z[i] = (_Float16)0.f;
  return z;
}

// A-fragment (16x32 f16) from an LDS tile stored row-major [16][stride_h].
// Lane L: m = L%16, half = L/16. elems 0..7 -> K = k0+half*8+i,
// elems 8..15 -> K = k0+16+half*8+(i-8). Two 16B ds reads.
DEVINL v16h load_afrag(const _Float16* base, int stride_h, int lane, int k0) {
  const int m  = lane & 15;
  const int hf = lane >> 4;
  const _Float16* p = base + m * stride_h + k0 + hf * 8;
  v8h lo = *(const v8h*)(p);
  v8h hi = *(const v8h*)(p + 16);
  return __builtin_shufflevector(lo, hi, 0, 1, 2, 3, 4, 5, 6, 7,
                                 8, 9, 10, 11, 12, 13, 14, 15);
}

// B-fragment (32x16 f16): packed[ktile][ntile][lane][16 f16] contiguous.
DEVINL v16h load_bfrag(const _Float16* wp, int kt, int nt, int lane) {
  const _Float16* p = wp + (((kt * 8 + nt) * 32 + lane) << 4);
  v8h lo = *(const v8h*)(p);
  v8h hi = *(const v8h*)(p + 8);
  return __builtin_shufflevector(lo, hi, 0, 1, 2, 3, 4, 5, 6, 7,
                                 8, 9, 10, 11, 12, 13, 14, 15);
}

// order-preserving float <-> uint for atomic max
DEVINL unsigned enc_f(float f) {
  unsigned u = __float_as_uint(f);
  return (u & 0x80000000u) ? ~u : (u | 0x80000000u);
}
DEVINL float dec_f(unsigned u) {
  return __uint_as_float((u & 0x80000000u) ? (u ^ 0x80000000u) : ~u);
}

// ---------------- weight repack: row-major f32 [K][128] -> B-fragment f16 ----------------

__global__ void pack_b_kernel(const float* __restrict__ W, _Float16* __restrict__ dst,
                              int K, int Ktiles) {
  int idx = blockIdx.x * blockDim.x + threadIdx.x;
  if (idx >= Ktiles * 8 * 32) return;
  int lane  = idx & 31;
  int ktile = idx >> 8;
  int n  = ((idx >> 5) & 7) * 16 + (lane & 15);
  int hf = lane >> 4;
  _Float16* p = dst + ((size_t)idx << 4);  // idx == ((ktile*8+ntile)*32+lane)
#pragma unroll
  for (int i = 0; i < 16; ++i) {
    int j  = i & 7;
    int kk = ktile * 32 + hf * 8 + j + ((i & 8) ? 16 : 0);
    float v = (kk < K) ? W[(size_t)kk * H + n] : 0.f;
    p[i] = (_Float16)v;
  }
}

// ---------------- embedding: h16 = f16(x @ emb_W + emb_b) ----------------

__launch_bounds__(128)
__global__ void embed_kernel(const float* __restrict__ x,
                             const _Float16* __restrict__ Wp,
                             const float* __restrict__ bias,
                             _Float16* __restrict__ hout) {
  __shared__ _Float16 sW[2 * 8 * 512];   // 16 KB
  __shared__ _Float16 sA[4][16 * 72];
  const int tid  = threadIdx.x;
  const int lane = tid & 31;
  const int wv   = tid >> 5;
  const int col  = lane & 15;
  const int hf   = lane >> 4;

  for (int i = tid; i < (2 * 8 * 512) / 8; i += 128)
    async_cp16(Wp + i * 8, sW + i * 8);
  async_wait0();
  __syncthreads();

  float bv[8];
#pragma unroll
  for (int nt = 0; nt < 8; ++nt) bv[nt] = bias[nt * 16 + col];

  for (int it = 0; it < 4; ++it) {
    int tile = (blockIdx.x * 4 + wv) * 4 + it;
    if (tile >= N_TILES) break;
    int nb = tile * 16;
    _Float16* A = sA[wv];
    for (int r = 0; r < 16; ++r) {
      int n = nb + r;
      _Float16* row = A + r * 72;
      int c = lane * 2;
      if (n < NN) {
        row[c]     = (_Float16)x[(size_t)n * ND + c];
        row[c + 1] = (_Float16)x[(size_t)n * ND + c + 1];
      } else {
        row[c] = (_Float16)0.f;
        row[c + 1] = (_Float16)0.f;
      }
      if (lane < 8) row[64 + lane] = (_Float16)0.f;
    }
    v8f acc[8];
#pragma unroll
    for (int nt = 0; nt < 8; ++nt) acc[nt] = zero8f();
#pragma unroll
    for (int kt = 0; kt < 2; ++kt) {
      v16h a = load_afrag(A, 72, lane, kt * 32);
#pragma unroll
      for (int nt = 0; nt < 8; ++nt)
        acc[nt] = wmma_f16(a, load_bfrag(sW, kt, nt, lane), acc[nt]);
    }
#pragma unroll
    for (int nt = 0; nt < 8; ++nt)
#pragma unroll
      for (int v = 0; v < 8; ++v) {
        int n = nb + v + 8 * hf;
        if (n < NN)
          hout[(size_t)n * H + nt * 16 + col] = (_Float16)(acc[nt][v] + bv[nt]);
      }
  }
}

// ---------------- edge message MLP + scatter-add ----------------
// m = relu([h[dst] | h[src] | edge_attr] @ W1 + b1) @ W2 + b2 ; aggr[dst] += m

__launch_bounds__(128)
__global__ void edge_msg_kernel(const _Float16* __restrict__ h,
                                const float* __restrict__ edge_attr,
                                const int* __restrict__ srcI,
                                const int* __restrict__ dstI,
                                const _Float16* __restrict__ W1p,
                                const _Float16* __restrict__ W2p,
                                const float* __restrict__ b1,
                                const float* __restrict__ b2,
                                float* __restrict__ aggr) {
  __shared__ _Float16 sW1[9 * 8 * 512];   // 72 KB packed W1 (K padded to 288)
  __shared__ _Float16 sW2[4 * 8 * 512];   // 32 KB packed W2
  __shared__ _Float16 sA[4][16 * 296];    // per-wave A tile, row stride 296
  const int tid  = threadIdx.x;
  const int lane = tid & 31;
  const int wv   = tid >> 5;
  const int col  = lane & 15;
  const int hf   = lane >> 4;

  // cooperative: stage this layer's packed weights into LDS once per WG
  for (int i = tid; i < (9 * 8 * 512) / 8; i += 128)
    async_cp16(W1p + i * 8, sW1 + i * 8);
  for (int i = tid; i < (4 * 8 * 512) / 8; i += 128)
    async_cp16(W2p + i * 8, sW2 + i * 8);
  async_wait0();
  __syncthreads();

  float bv1[8], bv2[8];
#pragma unroll
  for (int nt = 0; nt < 8; ++nt) {
    bv1[nt] = b1[nt * 16 + col];
    bv2[nt] = b2[nt * 16 + col];
  }

  for (int it = 0; it < 4; ++it) {
    int tile = (blockIdx.x * 4 + wv) * 4 + it;
    if (tile >= E_TILES) break;
    int eb = tile * 16;
    _Float16* A = sA[wv];

    // ---- stage A tile: [x_i | x_j | edge_attr | 0 pad] as f16 ----
    for (int r = 0; r < 16; ++r) {
      int e = eb + r;                      // uniform across the wave
      _Float16* row = A + r * 296;
      int c = lane * 8;                    // 8 f16 per lane covers cols 0..255
      if (e < NE) {
        int vd = dstI[e], vs = srcI[e];
        const _Float16* sp = (c < 128) ? (h + (size_t)vd * H + c)
                                       : (h + (size_t)vs * H + (c - 128));
        async_cp16(sp, row + c);           // gather 16B straight into LDS
        row[256 + lane] = (lane < ED) ? (_Float16)edge_attr[(size_t)e * ED + lane]
                                      : (_Float16)0.f;
      } else {
        *(v8h*)(row + c) = zero8h();
        row[256 + lane] = (_Float16)0.f;
      }
    }
    async_wait0();  // wave-private tile: wait own async copies only

    // ---- GEMM1: 16x288 @ 288x128 ----
    v8f acc[8];
#pragma unroll
    for (int nt = 0; nt < 8; ++nt) acc[nt] = zero8f();
    for (int kt = 0; kt < 9; ++kt) {
      v16h a = load_afrag(A, 296, lane, kt * 32);
#pragma unroll
      for (int nt = 0; nt < 8; ++nt)
        acc[nt] = wmma_f16(a, load_bfrag(sW1, kt, nt, lane), acc[nt]);
    }

    // ---- bias + relu, transpose C-layout -> A-layout via LDS (reuse tile) ----
#pragma unroll
    for (int nt = 0; nt < 8; ++nt)
#pragma unroll
      for (int v = 0; v < 8; ++v) {
        float xv = fmaxf(acc[nt][v] + bv1[nt], 0.f);
        A[(v + 8 * hf) * 136 + nt * 16 + col] = (_Float16)xv;
      }

    // ---- GEMM2: 16x128 @ 128x128 ----
#pragma unroll
    for (int nt = 0; nt < 8; ++nt) acc[nt] = zero8f();
#pragma unroll
    for (int kt = 0; kt < 4; ++kt) {
      v16h a = load_afrag(A, 136, lane, kt * 32);
#pragma unroll
      for (int nt = 0; nt < 8; ++nt)
        acc[nt] = wmma_f16(a, load_bfrag(sW2, kt, nt, lane), acc[nt]);
    }

    // ---- scatter-add m (+b2) into aggr[dst] ----
    int rowDst[8];
#pragma unroll
    for (int v = 0; v < 8; ++v) {
      int e = eb + v + 8 * hf;
      rowDst[v] = (e < NE) ? dstI[e] : -1;
    }
#pragma unroll
    for (int nt = 0; nt < 8; ++nt)
#pragma unroll
      for (int v = 0; v < 8; ++v)
        if (rowDst[v] >= 0)
          atomicAdd(aggr + (size_t)rowDst[v] * H + nt * 16 + col,
                    acc[nt][v] + bv2[nt]);
  }
}

// ---------------- node update MLP ----------------
// h' = relu( relu([h | aggr] @ U1 + c1) @ U2 + c2 )

__launch_bounds__(128)
__global__ void node_update_kernel(const _Float16* __restrict__ hin,
                                   const float* __restrict__ aggr,
                                   const _Float16* __restrict__ W1p,
                                   const _Float16* __restrict__ W2p,
                                   const float* __restrict__ b1,
                                   const float* __restrict__ b2,
                                   _Float16* __restrict__ hout) {
  __shared__ _Float16 sW1[8 * 8 * 512];   // 64 KB packed U1 (K=256)
  __shared__ _Float16 sW2[4 * 8 * 512];   // 32 KB packed U2
  __shared__ _Float16 sA[4][16 * 264];    // per-wave A tile, row stride 264
  const int tid  = threadIdx.x;
  const int lane = tid & 31;
  const int wv   = tid >> 5;
  const int col  = lane & 15;
  const int hf   = lane >> 4;

  for (int i = tid; i < (8 * 8 * 512) / 8; i += 128)
    async_cp16(W1p + i * 8, sW1 + i * 8);
  for (int i = tid; i < (4 * 8 * 512) / 8; i += 128)
    async_cp16(W2p + i * 8, sW2 + i * 8);
  async_wait0();
  __syncthreads();

  float bv1[8], bv2[8];
#pragma unroll
  for (int nt = 0; nt < 8; ++nt) {
    bv1[nt] = b1[nt * 16 + col];
    bv2[nt] = b2[nt * 16 + col];
  }

  for (int it = 0; it < 4; ++it) {
    int tile = (blockIdx.x * 4 + wv) * 4 + it;
    if (tile >= N_TILES) break;
    int nb = tile * 16;
    _Float16* A = sA[wv];

    // ---- stage [h16 | f16(aggr)] ----
    for (int r = 0; r < 16; ++r) {
      int n = nb + r;
      _Float16* row = A + r * 264;
      int c = lane * 4;
      if (n < NN) {
        *(v4h*)(row + c) = *(const v4h*)(hin + (size_t)n * H + c);
        float4 f = *(const float4*)(aggr + (size_t)n * H + c);
        v4h hv;
        hv[0] = (_Float16)f.x; hv[1] = (_Float16)f.y;
        hv[2] = (_Float16)f.z; hv[3] = (_Float16)f.w;
        *(v4h*)(row + 128 + c) = hv;
      } else {
        v4h z; z[0] = z[1] = z[2] = z[3] = (_Float16)0.f;
        *(v4h*)(row + c) = z;
        *(v4h*)(row + 128 + c) = z;
      }
    }

    // ---- GEMM1: 16x256 @ 256x128 ----
    v8f acc[8];
#pragma unroll
    for (int nt = 0; nt < 8; ++nt) acc[nt] = zero8f();
#pragma unroll
    for (int kt = 0; kt < 8; ++kt) {
      v16h a = load_afrag(A, 264, lane, kt * 32);
#pragma unroll
      for (int nt = 0; nt < 8; ++nt)
        acc[nt] = wmma_f16(a, load_bfrag(sW1, kt, nt, lane), acc[nt]);
    }

    // ---- bias+relu, transpose ----
#pragma unroll
    for (int nt = 0; nt < 8; ++nt)
#pragma unroll
      for (int v = 0; v < 8; ++v) {
        float xv = fmaxf(acc[nt][v] + bv1[nt], 0.f);
        A[(v + 8 * hf) * 136 + nt * 16 + col] = (_Float16)xv;
      }

    // ---- GEMM2: 16x128 @ 128x128 ----
#pragma unroll
    for (int nt = 0; nt < 8; ++nt) acc[nt] = zero8f();
#pragma unroll
    for (int kt = 0; kt < 4; ++kt) {
      v16h a = load_afrag(A, 136, lane, kt * 32);
#pragma unroll
      for (int nt = 0; nt < 8; ++nt)
        acc[nt] = wmma_f16(a, load_bfrag(sW2, kt, nt, lane), acc[nt]);
    }

    // ---- bias + final relu, store f16 ----
#pragma unroll
    for (int nt = 0; nt < 8; ++nt)
#pragma unroll
      for (int v = 0; v < 8; ++v) {
        int n = nb + v + 8 * hf;
        if (n < NN)
          hout[(size_t)n * H + nt * 16 + col] =
              (_Float16)fmaxf(acc[nt][v] + bv2[nt], 0.f);
      }
  }
}

// ---------------- pooling ----------------

__global__ void init_maxenc_kernel(unsigned* __restrict__ g_maxe) {
  int i = blockIdx.x * blockDim.x + threadIdx.x;
  if (i < NG * H) g_maxe[i] = MAXENC_INIT;
}

__global__ void pool_kernel(const _Float16* __restrict__ h,
                            const int* __restrict__ batch,
                            float* __restrict__ g_sum,
                            float* __restrict__ g_cnt,
                            unsigned* __restrict__ g_maxe) {
  int idx = blockIdx.x * blockDim.x + threadIdx.x;
  int n  = idx >> 5;
  int ln = idx & 31;
  if (n >= NN) return;
  int b = batch[n];
  if (ln == 0) atomicAdd(&g_cnt[b], 1.f);
#pragma unroll
  for (int j = 0; j < 4; ++j) {
    int c = ln * 4 + j;
    float v = (float)h[(size_t)n * H + c];
    atomicAdd(&g_sum[(size_t)b * H + c], v);
    atomicMax(&g_maxe[(size_t)b * H + c], enc_f(v));
  }
}

// ---------------- classifier (tiny; f32) ----------------

__launch_bounds__(128)
__global__ void classifier_kernel(const float* __restrict__ g_sum,
                                  const float* __restrict__ g_cnt,
                                  const unsigned* __restrict__ g_maxe,
                                  const float* __restrict__ W1, const float* __restrict__ b1,
                                  const float* __restrict__ W2, const float* __restrict__ b2,
                                  const float* __restrict__ W3, const float* __restrict__ b3,
                                  float* __restrict__ out) {
  __shared__ float rep[2 * H];
  __shared__ float z1[H];
  __shared__ float z2[H / 2];
  int g = blockIdx.x;
  int t = threadIdx.x;
  float denom = fmaxf(g_cnt[g], 1.f);
  rep[t] = g_sum[(size_t)g * H + t] / denom;
  unsigned me = g_maxe[(size_t)g * H + t];
  rep[H + t] = (me != MAXENC_INIT) ? dec_f(me) : 0.f;
  __syncthreads();
  float s = b1[t];
  for (int k = 0; k < 2 * H; ++k) s += rep[k] * W1[k * H + t];
  z1[t] = fmaxf(s, 0.f);
  __syncthreads();
  if (t < H / 2) {
    float s2 = b2[t];
    for (int k = 0; k < H; ++k) s2 += z1[k] * W2[k * (H / 2) + t];
    z2[t] = fmaxf(s2, 0.f);
  }
  __syncthreads();
  if (t < NC) {
    float s3 = b3[t];
    for (int k = 0; k < H / 2; ++k) s3 += z2[k] * W3[k * NC + t];
    out[g * NC + t] = s3;
  }
}

// ---------------- host launch ----------------

extern "C" void kernel_launch(void* const* d_in, const int* in_sizes, int n_in,
                              void* d_out, int out_size, void* d_ws, size_t ws_size,
                              hipStream_t stream) {
  (void)in_sizes; (void)n_in; (void)out_size; (void)ws_size;

  const float* x         = (const float*)d_in[0];
  const float* edge_attr = (const float*)d_in[1];
  const int*   edge_idx  = (const int*)d_in[2];
  const int*   batch     = (const int*)d_in[3];
  const float* emb_W     = (const float*)d_in[4];
  const float* emb_b     = (const float*)d_in[5];
  const float* msg_W1    = (const float*)d_in[6];
  const float* msg_b1    = (const float*)d_in[7];
  const float* msg_W2    = (const float*)d_in[8];
  const float* msg_b2    = (const float*)d_in[9];
  const float* upd_W1    = (const float*)d_in[10];
  const float* upd_b1    = (const float*)d_in[11];
  const float* upd_W2    = (const float*)d_in[12];
  const float* upd_b2    = (const float*)d_in[13];
  const float* cls_W1    = (const float*)d_in[14];
  const float* cls_b1    = (const float*)d_in[15];
  const float* cls_W2    = (const float*)d_in[16];
  const float* cls_b2    = (const float*)d_in[17];
  const float* cls_W3    = (const float*)d_in[18];
  const float* cls_b3    = (const float*)d_in[19];

  const int* srcI = edge_idx;         // edge_index[0] = src
  const int* dstI = edge_idx + NE;    // edge_index[1] = dst

  // ---- carve workspace (256B aligned segments) ----
  size_t off = 0;
  auto carve = [&](size_t bytes) -> void* {
    void* p = (char*)d_ws + off;
    off += (bytes + 255) & ~(size_t)255;
    return p;
  };
  _Float16* hA     = (_Float16*)carve((size_t)NN * H * 2);
  _Float16* hB     = (_Float16*)carve((size_t)NN * H * 2);
  float*    aggr   = (float*)carve((size_t)NN * H * 4);
  _Float16* embWp  = (_Float16*)carve((size_t)2 * 8 * 512 * 2);
  _Float16* msgW1p = (_Float16*)carve((size_t)NL * 9 * 8 * 512 * 2);
  _Float16* msgW2p = (_Float16*)carve((size_t)NL * 4 * 8 * 512 * 2);
  _Float16* updW1p = (_Float16*)carve((size_t)NL * 8 * 8 * 512 * 2);
  _Float16* updW2p = (_Float16*)carve((size_t)NL * 4 * 8 * 512 * 2);
  float*    g_sum  = (float*)carve((size_t)NG * H * 4);
  float*    g_cnt  = (float*)carve((size_t)NG * 4);
  unsigned* g_maxe = (unsigned*)carve((size_t)NG * H * 4);

  // ---- repack weights into WMMA-B fragment layout (f16) ----
  auto pack = [&](const float* W, _Float16* dst, int K, int Kt) {
    int total = Kt * 8 * 32;
    pack_b_kernel<<<(total + 255) / 256, 256, 0, stream>>>(W, dst, K, Kt);
  };
  pack(emb_W, embWp, ND, 2);
  for (int l = 0; l < NL; ++l) {
    pack(msg_W1 + (size_t)l * (2 * H + ED) * H, msgW1p + (size_t)l * 9 * 8 * 512, 2 * H + ED, 9);
    pack(msg_W2 + (size_t)l * H * H,            msgW2p + (size_t)l * 4 * 8 * 512, H, 4);
    pack(upd_W1 + (size_t)l * 2 * H * H,        updW1p + (size_t)l * 8 * 8 * 512, 2 * H, 8);
    pack(upd_W2 + (size_t)l * H * H,            updW2p + (size_t)l * 4 * 8 * 512, H, 4);
  }

  // ---- embedding ----
  {
    int grid = (N_TILES + 15) / 16;  // 4 waves x 4 tiles per WG
    embed_kernel<<<grid, 128, 0, stream>>>(x, embWp, emb_b, hA);
  }

  // ---- MPNN layers ----
  _Float16* hin = hA;
  _Float16* hout = hB;
  for (int l = 0; l < NL; ++l) {
    (void)hipMemsetAsync(aggr, 0, (size_t)NN * H * 4, stream);
    int egrid = (E_TILES + 15) / 16;
    edge_msg_kernel<<<egrid, 128, 0, stream>>>(
        hin, edge_attr, srcI, dstI,
        msgW1p + (size_t)l * 9 * 8 * 512, msgW2p + (size_t)l * 4 * 8 * 512,
        msg_b1 + (size_t)l * H, msg_b2 + (size_t)l * H, aggr);
    int ngrid = (N_TILES + 15) / 16;
    node_update_kernel<<<ngrid, 128, 0, stream>>>(
        hin, aggr,
        updW1p + (size_t)l * 8 * 8 * 512, updW2p + (size_t)l * 4 * 8 * 512,
        upd_b1 + (size_t)l * H, upd_b2 + (size_t)l * H, hout);
    _Float16* tmp = hin; hin = hout; hout = tmp;
  }

  // ---- pooling ----
  (void)hipMemsetAsync(g_sum, 0, (size_t)NG * H * 4, stream);
  (void)hipMemsetAsync(g_cnt, 0, (size_t)NG * 4, stream);
  init_maxenc_kernel<<<(NG * H + 255) / 256, 256, 0, stream>>>(g_maxe);
  pool_kernel<<<((size_t)NN * 32 + 255) / 256, 256, 0, stream>>>(hin, batch, g_sum, g_cnt, g_maxe);

  // ---- classifier ----
  classifier_kernel<<<NG, 128, 0, stream>>>(g_sum, g_cnt, g_maxe,
                                            cls_W1, cls_b1, cls_W2, cls_b2,
                                            cls_W3, cls_b3, (float*)d_out);
}
